// MultiHeadAttention_26276609916992
// MI455X (gfx1250) — matrix operand order repaired and checked by reading, not compiled, not run on hardware
//
#include <hip/hip_runtime.h>

// ---------------------------------------------------------------------------
// Problem constants (reference: B=4, T=2048, C=1024, H=16, D=64)
// ---------------------------------------------------------------------------
constexpr int NB = 4, NT = 2048, NC = 1024, NH = 16, ND = 64;
constexpr int NM = NB * NT;      // 8192 GEMM rows
constexpr int N3C = 3 * NC;      // 3072

// Workspace layout (bytes). Everything bf16 except final output.
constexpr size_t SZ_XB   = (size_t)NM * NC * 2;            // 16 MiB  x in bf16
constexpr size_t SZ_WQKV = (size_t)NC * N3C * 2;           //  6 MiB  W_qkv^T bf16
constexpr size_t SZ_WPRJ = (size_t)NC * NC * 2;            //  2 MiB  W_proj^T bf16
constexpr size_t SZ_HEAD = (size_t)NB * NH * NT * ND * 2;  // 16 MiB  per Q/K/Vt
constexpr size_t OFF_XB = 0;
constexpr size_t OFF_WQ = OFF_XB + SZ_XB;
constexpr size_t OFF_WP = OFF_WQ + SZ_WQKV;
constexpr size_t OFF_Q  = OFF_WP + SZ_WPRJ;
constexpr size_t OFF_K  = OFF_Q  + SZ_HEAD;
constexpr size_t OFF_V  = OFF_K  + SZ_HEAD;
constexpr size_t OFF_Y  = OFF_V  + SZ_HEAD;                // 16 MiB  attn out bf16

typedef __attribute__((ext_vector_type(4)))  __bf16 bf16x4;
typedef __attribute__((ext_vector_type(8)))  __bf16 bf16x8;
typedef __attribute__((ext_vector_type(16))) __bf16 bf16x16;
typedef __attribute__((ext_vector_type(8)))  float  f32x8;

__device__ __forceinline__ f32x8 zero_f32x8() {
  f32x8 z = {0.f, 0.f, 0.f, 0.f, 0.f, 0.f, 0.f, 0.f};
  return z;
}

__device__ __forceinline__ bf16x16 combine16(bf16x8 lo, bf16x8 hi) {
  bf16x16 r;
#pragma unroll
  for (int i = 0; i < 8; ++i) { r[i] = lo[i]; r[i + 8] = hi[i]; }
  return r;
}

// A fragment, 16x32 bf16 (row = lane&15 handled by caller via row pointer):
//   elements 0..7  <- K = lh*8 + 0..7
//   elements 8..15 <- K = 16 + lh*8 + 0..7        (lh = lane>>4)
// Row-major source => two contiguous 16B loads per lane.
__device__ __forceinline__ bf16x16 load_frag_a(const __bf16* rowk, int lh) {
  bf16x8 lo = *reinterpret_cast<const bf16x8*>(rowk + lh * 8);
  bf16x8 hi = *reinterpret_cast<const bf16x8*>(rowk + 16 + lh * 8);
  return combine16(lo, hi);
}

// B fragment, 32x16 bf16 (col n = lane&15 handled by caller via col pointer):
//   elements 0..7  <- K = lh*8 + 0..7
//   elements 8..15 <- K = 16 + lh*8 + 0..7        (lh = lane>>4)
// Source stored K-contiguous per column ([N][K]) => two contiguous 16B loads.
__device__ __forceinline__ bf16x16 load_frag_b(const __bf16* colk, int lh) {
  bf16x8 lo = *reinterpret_cast<const bf16x8*>(colk + lh * 8);
  bf16x8 hi = *reinterpret_cast<const bf16x8*>(colk + 16 + lh * 8);
  return combine16(lo, hi);
}

__device__ __forceinline__ f32x8 wmma_bf16(bf16x16 a, bf16x16 b, f32x8 c) {
  return __builtin_amdgcn_wmma_f32_16x16x32_bf16(false, a, false, b,
                                                 (short)0, c, false, false);
}

__device__ __forceinline__ float half16_max(float v) {
#pragma unroll
  for (int m = 1; m < 16; m <<= 1) v = fmaxf(v, __shfl_xor(v, m, 32));
  return v;
}
__device__ __forceinline__ float half16_sum(float v) {
#pragma unroll
  for (int m = 1; m < 16; m <<= 1) v += __shfl_xor(v, m, 32);
  return v;
}

// ---------------------------------------------------------------------------
// K1: x (fp32) -> bf16, vectorized
// ---------------------------------------------------------------------------
__global__ void cvt_x_bf16(const float* __restrict__ src,
                           __bf16* __restrict__ dst, int n4) {
  int i = blockIdx.x * blockDim.x + threadIdx.x;
  if (i >= n4) return;
  float4 v = reinterpret_cast<const float4*>(src)[i];
  bf16x4 o;
  o[0] = (__bf16)v.x; o[1] = (__bf16)v.y; o[2] = (__bf16)v.z; o[3] = (__bf16)v.w;
  reinterpret_cast<bf16x4*>(dst)[i] = o;
}

// ---------------------------------------------------------------------------
// K0: W [K][N] fp32 -> Wt [N][K] bf16 (transpose so B-fragments load 16B runs)
// ---------------------------------------------------------------------------
__global__ void cvt_w_transpose(const float* __restrict__ W,
                                __bf16* __restrict__ Wt, int K, int N) {
  int idx = blockIdx.x * blockDim.x + threadIdx.x;
  if (idx >= K * N) return;
  int n = idx / K, k = idx % K;
  Wt[idx] = (__bf16)W[(size_t)k * N + n];
}

// ---------------------------------------------------------------------------
// K2: QKV GEMM  qkv[m][n] = sum_k Xb[m][k] * Wqkv_t[n][k]
// 256 threads = 8 waves; block tile 128(M) x 128(N); wave tile 64(M) x 32(N)
// => 4x2 accumulator tiles per wave, 8 wmma per k-step, K-loop step 32.
// Output scattered straight into head-major Q [B,H,T,D] (pre-scaled by
// 1/sqrt(D)), K [B,H,T,D], and V transposed [B,H,D,T] for the PV B-fragments.
// ---------------------------------------------------------------------------
__global__ __launch_bounds__(256) void qkv_gemm(
    const __bf16* __restrict__ Xb, const __bf16* __restrict__ Wt,
    __bf16* __restrict__ Qs, __bf16* __restrict__ Ks, __bf16* __restrict__ Vts) {
  const int lane = threadIdx.x & 31;
  const int w    = threadIdx.x >> 5;
  const int lh = lane >> 4, lr = lane & 15;
  const int bm = blockIdx.y * 128 + (w >> 2) * 64;   // wave M base
  const int bn = blockIdx.x * 128 + (w & 3) * 32;    // wave N base

  f32x8 acc[4][2];
#pragma unroll
  for (int mt = 0; mt < 4; ++mt)
#pragma unroll
    for (int nt = 0; nt < 2; ++nt) acc[mt][nt] = zero_f32x8();

  for (int k0 = 0; k0 < NC; k0 += 32) {
    // Speculative WGP-scope prefetch of next k-step (no branch, OOB is safe).
    __builtin_prefetch(Xb + (size_t)(bm + lr) * NC + k0 + 32, 0, 3);
    __builtin_prefetch(Wt + (size_t)(bn + lr) * NC + k0 + 32, 0, 3);
    bf16x16 afr[4], bfr[2];
#pragma unroll
    for (int mt = 0; mt < 4; ++mt)
      afr[mt] = load_frag_a(Xb + (size_t)(bm + mt * 16 + lr) * NC + k0, lh);
#pragma unroll
    for (int nt = 0; nt < 2; ++nt)
      bfr[nt] = load_frag_b(Wt + (size_t)(bn + nt * 16 + lr) * NC + k0, lh);
#pragma unroll
    for (int mt = 0; mt < 4; ++mt)
#pragma unroll
      for (int nt = 0; nt < 2; ++nt)
        acc[mt][nt] = wmma_bf16(afr[mt], bfr[nt], acc[mt][nt]);
  }

#pragma unroll
  for (int mt = 0; mt < 4; ++mt)
#pragma unroll
    for (int nt = 0; nt < 2; ++nt)
#pragma unroll
      for (int r = 0; r < 8; ++r) {
        unsigned m = bm + mt * 16 + lh * 8 + r;
        unsigned n = bn + nt * 16 + lr;
        float v = acc[mt][nt][r];
        unsigned mat = n >> 10;            // 0=Q 1=K 2=V
        unsigned c = n & 1023u;
        unsigned h = c >> 6, d = c & 63u;
        unsigned b = m >> 11, t = m & 2047u;
        size_t bh = (size_t)b * NH + h;
        if (mat == 0)
          Qs[(bh * NT + t) * ND + d] = (__bf16)(v * 0.125f);  // fold 1/sqrt(64)
        else if (mat == 1)
          Ks[(bh * NT + t) * ND + d] = (__bf16)v;
        else
          Vts[(bh * ND + d) * NT + t] = (__bf16)v;            // transposed
      }
}

// ---------------------------------------------------------------------------
// K3: causal flash attention, one wave32 per 32-query tile per (b,h).
// Two 16-row query subtiles share every K and V fragment (2x reuse):
//   S  = Q(32x64) K^T  -> 8 wmma per 32-key step
//   O += P(32x32) V    -> 8 wmma per 32-key step
// Online softmax state lives in C-layout lanes; P relayout via 2.5KB LDS.
// ---------------------------------------------------------------------------
__global__ __launch_bounds__(32) void flash_attn(
    const __bf16* __restrict__ Q, const __bf16* __restrict__ K,
    const __bf16* __restrict__ Vt, __bf16* __restrict__ Yb) {
  __shared__ __bf16 Pl[2][16][40];  // per-subtile 16x32, padded to 40

  const int lane = threadIdx.x & 31;
  const int lh = lane >> 4, lr = lane & 15;
  const int q0 = blockIdx.x * 32, h = blockIdx.y, b = blockIdx.z;
  const size_t bh = (size_t)b * NH + h;
  const __bf16* Qh = Q  + bh * NT * ND;
  const __bf16* Kh = K  + bh * NT * ND;
  const __bf16* Vh = Vt + bh * ND * NT;

  bf16x16 qa[2][2];
#pragma unroll
  for (int qs = 0; qs < 2; ++qs) {
    qa[qs][0] = load_frag_a(Qh + (size_t)(q0 + qs * 16 + lr) * ND + 0,  lh);
    qa[qs][1] = load_frag_a(Qh + (size_t)(q0 + qs * 16 + lr) * ND + 32, lh);
  }

  f32x8 o[2][4];
  float mrow[2][8], lrow[2][8];
#pragma unroll
  for (int qs = 0; qs < 2; ++qs) {
#pragma unroll
    for (int dt = 0; dt < 4; ++dt) o[qs][dt] = zero_f32x8();
#pragma unroll
    for (int r = 0; r < 8; ++r) { mrow[qs][r] = -1e30f; lrow[qs][r] = 0.f; }
  }

  const int ntiles = (q0 >> 5) + 1;  // keys 0 .. q0+31 in tiles of 32
  for (int it = 0; it < ntiles; ++it) {
    const int j0 = it * 32;

    // K fragments loaded once, reused by both query subtiles.
    bf16x16 kb[2][2];
#pragma unroll
    for (int ti = 0; ti < 2; ++ti) {
      kb[ti][0] = load_frag_b(Kh + (size_t)(j0 + ti * 16 + lr) * ND + 0,  lh);
      kb[ti][1] = load_frag_b(Kh + (size_t)(j0 + ti * 16 + lr) * ND + 32, lh);
    }

    f32x8 s[2][2];
#pragma unroll
    for (int qs = 0; qs < 2; ++qs)
#pragma unroll
      for (int ti = 0; ti < 2; ++ti) {
        s[qs][ti] = zero_f32x8();
        s[qs][ti] = wmma_bf16(qa[qs][0], kb[ti][0], s[qs][ti]);
        s[qs][ti] = wmma_bf16(qa[qs][1], kb[ti][1], s[qs][ti]);
      }

    if (j0 + 31 > q0) {  // causal boundary tile(s)
#pragma unroll
      for (int qs = 0; qs < 2; ++qs)
#pragma unroll
        for (int ti = 0; ti < 2; ++ti)
#pragma unroll
          for (int r = 0; r < 8; ++r) {
            int key = j0 + ti * 16 + lr;
            int qr  = q0 + qs * 16 + lh * 8 + r;
            if (key > qr) s[qs][ti][r] = -1e30f;
          }
    }

#pragma unroll
    for (int qs = 0; qs < 2; ++qs) {
      float alpha[8];
#pragma unroll
      for (int r = 0; r < 8; ++r) {
        float mx = half16_max(fmaxf(s[qs][0][r], s[qs][1][r]));
        float mn = fmaxf(mrow[qs][r], mx);
        alpha[r] = __expf(mrow[qs][r] - mn);
        mrow[qs][r] = mn;
      }
#pragma unroll
      for (int r = 0; r < 8; ++r) {
        float p0 = __expf(s[qs][0][r] - mrow[qs][r]);
        float p1 = __expf(s[qs][1][r] - mrow[qs][r]);
        s[qs][0][r] = p0; s[qs][1][r] = p1;
        lrow[qs][r] = lrow[qs][r] * alpha[r] + half16_sum(p0 + p1);
      }
#pragma unroll
      for (int dt = 0; dt < 4; ++dt)
#pragma unroll
        for (int r = 0; r < 8; ++r) o[qs][dt][r] *= alpha[r];

      // P: C-layout -> LDS (per subtile)
#pragma unroll
      for (int ti = 0; ti < 2; ++ti)
#pragma unroll
        for (int r = 0; r < 8; ++r)
          Pl[qs][lh * 8 + r][ti * 16 + lr] = (__bf16)s[qs][ti][r];
    }
    __syncthreads();

    bf16x16 pa[2];
#pragma unroll
    for (int qs = 0; qs < 2; ++qs) pa[qs] = load_frag_a(&Pl[qs][lr][0], lh);

    // V fragments loaded once, reused by both query subtiles.
#pragma unroll
    for (int dt = 0; dt < 4; ++dt) {
      bf16x16 vb = load_frag_b(Vh + (size_t)(dt * 16 + lr) * NT + j0, lh);
#pragma unroll
      for (int qs = 0; qs < 2; ++qs)
        o[qs][dt] = wmma_bf16(pa[qs], vb, o[qs][dt]);
    }
    __syncthreads();
  }

#pragma unroll
  for (int qs = 0; qs < 2; ++qs)
#pragma unroll
    for (int r = 0; r < 8; ++r) {
      float inv = 1.0f / lrow[qs][r];
      int t = q0 + qs * 16 + lh * 8 + r;
#pragma unroll
      for (int dt = 0; dt < 4; ++dt)
        Yb[((size_t)b * NT + t) * NC + h * ND + dt * 16 + lr] =
            (__bf16)(o[qs][dt][r] * inv);
    }
}

// ---------------------------------------------------------------------------
// K4: projection  out[m][n] = sum_k Yb[m][k] * Wproj_t[n][k] + b[n]  (fp32 out)
// ---------------------------------------------------------------------------
__global__ __launch_bounds__(256) void proj_gemm(
    const __bf16* __restrict__ Yb, const __bf16* __restrict__ Wt,
    const float* __restrict__ bias, float* __restrict__ out) {
  const int lane = threadIdx.x & 31;
  const int w    = threadIdx.x >> 5;
  const int lh = lane >> 4, lr = lane & 15;
  const int bm = blockIdx.y * 128 + (w >> 2) * 64;
  const int bn = blockIdx.x * 128 + (w & 3) * 32;

  f32x8 acc[4][2];
#pragma unroll
  for (int mt = 0; mt < 4; ++mt)
#pragma unroll
    for (int nt = 0; nt < 2; ++nt) acc[mt][nt] = zero_f32x8();

  for (int k0 = 0; k0 < NC; k0 += 32) {
    __builtin_prefetch(Yb + (size_t)(bm + lr) * NC + k0 + 32, 0, 3);
    __builtin_prefetch(Wt + (size_t)(bn + lr) * NC + k0 + 32, 0, 3);
    bf16x16 afr[4], bfr[2];
#pragma unroll
    for (int mt = 0; mt < 4; ++mt)
      afr[mt] = load_frag_a(Yb + (size_t)(bm + mt * 16 + lr) * NC + k0, lh);
#pragma unroll
    for (int nt = 0; nt < 2; ++nt)
      bfr[nt] = load_frag_b(Wt + (size_t)(bn + nt * 16 + lr) * NC + k0, lh);
#pragma unroll
    for (int mt = 0; mt < 4; ++mt)
#pragma unroll
      for (int nt = 0; nt < 2; ++nt)
        acc[mt][nt] = wmma_bf16(afr[mt], bfr[nt], acc[mt][nt]);
  }

#pragma unroll
  for (int mt = 0; mt < 4; ++mt)
#pragma unroll
    for (int nt = 0; nt < 2; ++nt)
#pragma unroll
      for (int r = 0; r < 8; ++r) {
        int m = bm + mt * 16 + lh * 8 + r;
        int n = bn + nt * 16 + lr;
        out[(size_t)m * NC + n] = acc[mt][nt][r] + bias[n];
      }
}

// ---------------------------------------------------------------------------
extern "C" void kernel_launch(void* const* d_in, const int* in_sizes, int n_in,
                              void* d_out, int out_size, void* d_ws,
                              size_t ws_size, hipStream_t stream) {
  (void)in_sizes; (void)n_in; (void)out_size; (void)ws_size;
  const float* x     = (const float*)d_in[0];
  const float* Wqkv  = (const float*)d_in[1];
  const float* Wproj = (const float*)d_in[2];
  const float* bproj = (const float*)d_in[3];
  float* out = (float*)d_out;

  char* ws = (char*)d_ws;
  __bf16* Xb  = (__bf16*)(ws + OFF_XB);
  __bf16* Wqt = (__bf16*)(ws + OFF_WQ);
  __bf16* Wpt = (__bf16*)(ws + OFF_WP);
  __bf16* Qs  = (__bf16*)(ws + OFF_Q);
  __bf16* Ks  = (__bf16*)(ws + OFF_K);
  __bf16* Vts = (__bf16*)(ws + OFF_V);
  __bf16* Yb  = (__bf16*)(ws + OFF_Y);

  const int n4 = NM * NC / 4;
  cvt_x_bf16<<<n4 / 256, 256, 0, stream>>>(x, Xb, n4);
  cvt_w_transpose<<<(NC * N3C + 255) / 256, 256, 0, stream>>>(Wqkv, Wqt, NC, N3C);
  cvt_w_transpose<<<(NC * NC + 255) / 256, 256, 0, stream>>>(Wproj, Wpt, NC, NC);
  qkv_gemm<<<dim3(N3C / 128, NM / 128), 256, 0, stream>>>(Xb, Wqt, Qs, Ks, Vts);
  flash_attn<<<dim3(NT / 32, NH, NB), 32, 0, stream>>>(Qs, Ks, Vts, Yb);
  proj_gemm<<<dim3(NC / 128, NM / 128), 256, 0, stream>>>(Yb, Wpt, bproj, out);
}
